// NumericalMarkowitz_59081570124227
// MI455X (gfx1250) — matrix operand order repaired and checked by reading, not compiled, not run on hardware
//
#include <hip/hip_runtime.h>
#include <hip/hip_bf16.h>

#define N 256
#define CT_STRIDE 264            // halfs; 528 B row, 16B-aligned chunks
#define Q_STRIDE  264
#define CT_OFF 0
#define Q_OFF  (N * CT_STRIDE * 2)            // 135168
#define ZB_OFF (Q_OFF + N * Q_STRIDE * 2)     // 270336
#define VB_OFF (ZB_OFF + N * 4)               // 271360
#define RED_OFF (VB_OFF + N * 4)              // 272384
#define SMEM_BYTES (RED_OFF + 32)             // 272416 <= 320 KB WGP LDS

typedef __attribute__((ext_vector_type(16))) _Float16 v16h;
typedef __attribute__((ext_vector_type(8)))  _Float16 v8h;
typedef __attribute__((ext_vector_type(8)))  float    v8f;

// ---- wave32 reductions (all lanes end with the full result) ----
__device__ __forceinline__ float wsum(float x) {
  #pragma unroll
  for (int m = 16; m >= 1; m >>= 1) x += __shfl_xor(x, m, 32);
  return x;
}
__device__ __forceinline__ float wminr(float x) {
  #pragma unroll
  for (int m = 16; m >= 1; m >>= 1) x = fminf(x, __shfl_xor(x, m, 32));
  return x;
}
__device__ __forceinline__ float wmaxr(float x) {
  #pragma unroll
  for (int m = 16; m >= 1; m >>= 1) x = fmaxf(x, __shfl_xor(x, m, 32));
  return x;
}

// dot of one f16 Q row (LDS) with an f32 vector (LDS); mixed f16*f32 FMAs
__device__ __forceinline__ float dotrow(const _Float16* __restrict__ qrow,
                                        const float* __restrict__ x) {
  float acc = 0.0f;
  #pragma unroll 4
  for (int j = 0; j < N; j += 8) {
    v8h q = *(const v8h*)(qrow + j);
    #pragma unroll
    for (int t = 0; t < 8; ++t) acc = fmaf((float)q[t], x[j + t], acc);
  }
  return acc;
}

extern "C" __global__ void __launch_bounds__(256, 1)
markowitz_fused(const float* __restrict__ rets, const float* __restrict__ cov,
                const float* __restrict__ gsq,  const float* __restrict__ alp,
                float* __restrict__ out) {
  extern __shared__ char smem[];
  _Float16* ct  = (_Float16*)(smem + CT_OFF);   // C^T, gamma folded, f16
  _Float16* qm  = (_Float16*)(smem + Q_OFF);    // Q, f16
  float*    zb  = (float*)(smem + ZB_OFF);      // power-b / FISTA z
  float*    vb  = (float*)(smem + VB_OFF);      // FISTA pre-projection v
  float*    red = (float*)(smem + RED_OFF);     // 8 per-wave partials

  const int b    = blockIdx.x;
  const int tid  = threadIdx.x;
  const int wave = tid >> 5;
  const int lane = tid & 31;

  const float g   = gsq[b];
  const float aab = fabsf(alp[b]);
  const float rr  = rets[(size_t)b * N + tid];
  const float* cb = cov + (size_t)b * N * N;

  // ---------- Phase A: load gamma*C, transpose into LDS as f16 ----------
  for (int base = tid * 4; base < N * N; base += 256 * 4) {
    int k = base >> 8, i = base & (N - 1);
    float4 c4 = *reinterpret_cast<const float4*>(cb + base);
    ct[(i + 0) * CT_STRIDE + k] = (_Float16)(g * c4.x);
    ct[(i + 1) * CT_STRIDE + k] = (_Float16)(g * c4.y);
    ct[(i + 2) * CT_STRIDE + k] = (_Float16)(g * c4.z);
    ct[(i + 3) * CT_STRIDE + k] = (_Float16)(g * c4.w);
  }
  __syncthreads();

  // ---------- Phase B: Q = Ct * Ct^T via v_wmma_f32_16x16x32_f16 ----------
  // A frag: lane m holds row (16*ti + m%16); K-chunks [kb, kb+8) and [kb+16, kb+24), kb = 8*(lane>=16)
  // B frag: lane q holds column (q%16) of B == row (16*tj + q%16) of Ct; K-chunks [kb, kb+8) and [kb+16, kb+24), kb = 8*(lane>=16)
  {
    const int rA    = lane & 15;
    const int kAoff = (lane >> 4) << 3;
    const int kBoff = (lane >> 4) << 4;
    for (int tile = wave; tile < 256; tile += 8) {
      const int ti = tile >> 4, tj = tile & 15;
      const _Float16* arow = ct + (ti * 16 + rA) * CT_STRIDE;
      const _Float16* brow = ct + (tj * 16 + rA) * CT_STRIDE;
      v8f acc = {};
      #pragma unroll
      for (int kc = 0; kc < N; kc += 32) {
        union { v16h v; v8h h[2]; } A, B;
        A.h[0] = *(const v8h*)(arow + kc + kAoff);
        A.h[1] = *(const v8h*)(arow + kc + kAoff + 16);
        B.h[0] = *(const v8h*)(brow + kc + kAoff);
        B.h[1] = *(const v8h*)(brow + kc + kAoff + 16);
        acc = __builtin_amdgcn_wmma_f32_16x16x32_f16(
            false, A.v, false, B.v, (short)0, acc, false, false);
      }
      const int jc    = tj * 16 + (lane & 15);
      const int ibase = ti * 16 + ((lane >> 4) << 3);
      #pragma unroll
      for (int r = 0; r < 8; ++r) {
        int i = ibase + r;
        float val = acc[r];
        if (i == jc) val += aab;                 // + |alpha| * I
        qm[i * Q_STRIDE + jc] = (_Float16)val;
      }
    }
  }
  __syncthreads();

  const _Float16* qrow = qm + tid * Q_STRIDE;    // this thread owns row `tid`

  // ---------- Phase C: power iteration for step size ----------
  zb[tid] = 1.0f;
  __syncthreads();
  for (int it = 0; it < 30; ++it) {
    float p = dotrow(qrow, zb);
    float s = wsum(p * p);
    if (lane == 0) red[wave] = s;
    __syncthreads();
    float n2 = 0.0f;
    #pragma unroll
    for (int w = 0; w < 8; ++w) n2 += red[w];
    zb[tid] = p / (sqrtf(n2) + 1e-12f);
    __syncthreads();
  }
  float step;
  {
    float p = dotrow(qrow, zb);
    float s = wsum(zb[tid] * p);
    if (lane == 0) red[wave] = s;
    __syncthreads();
    float lam = 0.0f;
    #pragma unroll
    for (int w = 0; w < 8; ++w) lam += red[w];
    step = 1.0f / (2.0f * lam + 1e-6f);
  }

  // ---------- Phase D: FISTA (300 iters) ----------
  zb[tid] = 1.0f / N;
  float wprev[8];
  #pragma unroll
  for (int j = 0; j < 8; ++j) wprev[j] = 1.0f / N;
  float t = 1.0f;
  __syncthreads();

  for (int it = 0; it < 300; ++it) {
    // distributed: grad row per thread, v into LDS
    float gr = 2.0f * dotrow(qrow, zb) - rr;
    vb[tid] = zb[tid] - step * gr;
    __syncthreads();

    // replicated per-wave: full 256-vector, 8 elems/lane; wave-local reductions only
    float v[8];
    #pragma unroll
    for (int j = 0; j < 8; ++j) v[j] = vb[lane * 8 + j];

    float mn = v[0], mx = v[0];
    #pragma unroll
    for (int j = 1; j < 8; ++j) { mn = fminf(mn, v[j]); mx = fmaxf(mx, v[j]); }
    float lo = wminr(mn) - 1.0f;   // u = MAX_WEIGHT = 1
    float hi = wmaxr(mx);

    for (int bi = 0; bi < 60; ++bi) {
      float mid = 0.5f * (lo + hi);
      float sl = 0.0f;
      #pragma unroll
      for (int j = 0; j < 8; ++j) sl += fminf(fmaxf(v[j] - mid, 0.0f), 1.0f);
      float s = wsum(sl);
      if (s > 1.0f) lo = mid; else hi = mid;
    }
    float tau0 = 0.5f * (lo + hi);

    float cl = 0.0f, fl = 0.0f;
    #pragma unroll
    for (int j = 0; j < 8; ++j) {
      float d = v[j] - tau0;
      cl += (d > 0.0f && d < 1.0f) ? 1.0f : 0.0f;
      fl += fminf(fmaxf(d, 0.0f), 1.0f);
    }
    float kk  = fmaxf(wsum(cl), 1.0f);
    float f   = wsum(fl) - 1.0f;
    float tau = tau0 + f / kk;     // Newton step, exact as in reference

    float tn   = 0.5f * (1.0f + sqrtf(1.0f + 4.0f * t * t));
    float coef = (t - 1.0f) / tn;
    t = tn;
    #pragma unroll
    for (int j = 0; j < 8; ++j) {
      float wn = fminf(fmaxf(v[j] - tau, 0.0f), 1.0f);
      float zn = wn + coef * (wn - wprev[j]);
      wprev[j] = wn;
      if (wave == 0) zb[lane * 8 + j] = zn;   // one wave publishes z for next matvec
    }
    __syncthreads();
  }

  if (wave == 0) {
    #pragma unroll
    for (int j = 0; j < 8; ++j)
      out[(size_t)b * N + lane * 8 + j] = wprev[j];
  }
}

extern "C" void kernel_launch(void* const* d_in, const int* in_sizes, int n_in,
                              void* d_out, int out_size, void* d_ws, size_t ws_size,
                              hipStream_t stream) {
  const float* rets = (const float*)d_in[0];   // (512,256) f32
  const float* cov  = (const float*)d_in[1];   // (512,256,256) f32
  const float* gsq  = (const float*)d_in[2];   // (512,) f32
  const float* alp  = (const float*)d_in[3];   // (512,) f32
  float* out = (float*)d_out;                  // (512,256) f32
  markowitz_fused<<<dim3(512), dim3(256), SMEM_BYTES, stream>>>(rets, cov, gsq, alp, out);
}
